// Decoder_78176994721983
// MI455X (gfx1250) — compile-verified, matrix operand order checked by reading, not run on hardware
//
#include <hip/hip_runtime.h>
#include <hip/hip_bf16.h>
#include <stdint.h>

// ---------------------------------------------------------------------------
// CDNA5 (gfx1250) EGNN decoder.
//  - GEMMs: v_wmma_f32_16x16x32_bf16, 128x64x32 tiles, 8 wave32 waves/block.
//  - Weights pre-transposed/converted once to bf16 Wt[N][KP] (KP = ceil32(K),
//    zero padded) so B-tile staging is contiguous 16B chunks.
//  - Tile staging of bf16 operands uses GLOBAL_LOAD_ASYNC_TO_LDS_B128
//    (ASYNCcnt) via inline asm; s_wait_asynccnt 0 before the block barrier.
//  - LDS rows padded to 40 ushorts (80B): b128 frag loads stay 16B aligned
//    and bank-conflict-free (20*m mod 64 distinct for m=0..15).
// ---------------------------------------------------------------------------

typedef __attribute__((ext_vector_type(16))) __bf16 v16bf;
typedef __attribute__((ext_vector_type(8)))  float  v8f;

#define BM 128
#define BN 64
#define BK 32
#define LDR (BK + 8)   // 40 ushorts -> 80B row stride

__device__ __forceinline__ unsigned short f2bf(float x) {
    union { float f; uint32_t u; } c; c.f = x;
    uint32_t u = c.u;
    return (unsigned short)((u + 0x7FFFu + ((u >> 16) & 1u)) >> 16);  // RNE
}
__device__ __forceinline__ float bf2f(unsigned short b) {
    union { uint32_t u; float f; } c; c.u = ((uint32_t)b) << 16;
    return c.f;
}
__device__ __forceinline__ float silu_f(float x) {
    return x / (1.0f + __expf(-x));
}

// LDS offset = low 32 bits of a generic pointer into shared memory.
__device__ __forceinline__ unsigned ldsoff(const void* p) {
    return (unsigned)(uintptr_t)p;
}
// CDNA5 async global->LDS 16B copy (tracked by ASYNCcnt).
__device__ __forceinline__ void async_ld_b128(unsigned lds_dst, const void* gsrc) {
    asm volatile("global_load_async_to_lds_b128 %0, %1, off"
                 :: "v"(lds_dst), "v"(gsrc) : "memory");
}
__device__ __forceinline__ void wait_async0() {
    asm volatile("s_wait_asynccnt 0" ::: "memory");
}

union Frag { v16bf v; uint4 q[2]; unsigned short u[16]; };

__device__ __forceinline__ v8f wmma_bf16(const Frag& a, const Frag& b, v8f c) {
    return __builtin_amdgcn_wmma_f32_16x16x32_bf16(false, a.v, false, b.v,
                                                   (short)0, c, false, false);
}

// ---------------------------------------------------------------------------
// Weight prep: Wt[n*KP + k] = bf16(W[k*N + n]), zero pad k in [K, KP)
// ---------------------------------------------------------------------------
__global__ void xpose_w_kernel(const float* __restrict__ W,
                               unsigned short* __restrict__ Wt,
                               int K, int N, int KP) {
    int i = blockIdx.x * blockDim.x + threadIdx.x;
    if (i >= N * KP) return;
    int n = i / KP, k = i % KP;
    Wt[i] = (k < K) ? f2bf(W[(size_t)k * N + n]) : (unsigned short)0;
}

// ---------------------------------------------------------------------------
// Generic tiled GEMM: out = act(A[M,K] @ W[K,N] + bias)
// Wt is bf16 [N][KP]. ABF: 0 -> A f32; 1 -> A bf16 (K%8==0 shapes only).
// ---------------------------------------------------------------------------
template <int ABF>
__global__ __launch_bounds__(256) void gemm_kernel(
    const void* __restrict__ Aptr,
    const unsigned short* __restrict__ Wt, int KP,
    const float* __restrict__ bias,
    float* __restrict__ outF, unsigned short* __restrict__ outB,
    int M, int N, int K, int act)
{
    __shared__ __align__(16) unsigned short sA[BM][LDR];
    __shared__ __align__(16) unsigned short sBt[BN][LDR];

    const int tid  = threadIdx.x;
    const int lane = tid & 31;
    const int wv   = tid >> 5;
    const int wr   = wv & 3;
    const int wc   = wv >> 2;
    const int row0 = blockIdx.y * BM;
    const int col0 = blockIdx.x * BN;
    const int mm   = lane & 15;
    const int half = lane >> 4;

    // staging coords
    const int ar  = tid >> 1;          // A row 0..127
    const int akb = (tid & 1) << 4;    // A k-chunk 0/16
    const int bn  = tid >> 2;          // B col 0..63
    const int bkc = (tid & 3) << 3;    // B k-chunk 0/8/16/24

    v8f c00 = {}, c01 = {}, c10 = {}, c11 = {};

    const float*          Af = (const float*)Aptr;
    const unsigned short* Ab = (const unsigned short*)Aptr;

    for (int k0 = 0; k0 < K; k0 += BK) {
        // ---- stage A (128x32) ----
        {
            const int gr = row0 + ar;
            if (ABF) {
                // bf16 A: two async 16B global->LDS copies (always in-bounds
                // for our call sites: M%128==0, K%32==0)
                const unsigned short* s = Ab + (size_t)gr * K + k0 + akb;
                async_ld_b128(ldsoff(&sA[ar][akb]),     s);
                async_ld_b128(ldsoff(&sA[ar][akb + 8]), s + 8);
            } else {
                Frag p;
                if (gr < M && k0 + BK <= K) {
                    const float* s = Af + (size_t)gr * K + k0 + akb;
#pragma unroll
                    for (int j = 0; j < 16; ++j) p.u[j] = f2bf(s[j]);
                } else {
#pragma unroll
                    for (int j = 0; j < 16; ++j) {
                        const int gk = k0 + akb + j;
                        p.u[j] = (gr < M && gk < K)
                                 ? f2bf(Af[(size_t)gr * K + gk]) : (unsigned short)0;
                    }
                }
                *(uint4*)&sA[ar][akb]     = p.q[0];
                *(uint4*)&sA[ar][akb + 8] = p.q[1];
            }
        }
        // ---- stage B (64 cols x 32 k) from pre-transposed bf16 weights ----
        {
            const int gn = col0 + bn;
            if (gn < N) {
                async_ld_b128(ldsoff(&sBt[bn][bkc]),
                              Wt + (size_t)gn * KP + k0 + bkc);
            } else {
                uint4 zz; zz.x = zz.y = zz.z = zz.w = 0u;
                *(uint4*)&sBt[bn][bkc] = zz;
            }
        }
        wait_async0();
        __syncthreads();

        // ---- fragments: two ds_load_b128 each ----
        Frag a0, a1, b0, b1;
        a0.q[0] = *(const uint4*)&sA[wr * 32 + mm][8 * half];
        a0.q[1] = *(const uint4*)&sA[wr * 32 + mm][16 + 8 * half];
        a1.q[0] = *(const uint4*)&sA[wr * 32 + 16 + mm][8 * half];
        a1.q[1] = *(const uint4*)&sA[wr * 32 + 16 + mm][16 + 8 * half];
        b0.q[0] = *(const uint4*)&sBt[wc * 32 + mm][8 * half];
        b0.q[1] = *(const uint4*)&sBt[wc * 32 + mm][16 + 8 * half];
        b1.q[0] = *(const uint4*)&sBt[wc * 32 + 16 + mm][8 * half];
        b1.q[1] = *(const uint4*)&sBt[wc * 32 + 16 + mm][16 + 8 * half];

        c00 = wmma_bf16(a0, b0, c00);
        c01 = wmma_bf16(a0, b1, c01);
        c10 = wmma_bf16(a1, b0, c10);
        c11 = wmma_bf16(a1, b1, c11);
        __syncthreads();
    }

    auto store_acc = [&](const v8f& acc, int rbase, int cbase) {
#pragma unroll
        for (int i = 0; i < 8; ++i) {
            const int r = rbase + half * 8 + i;
            const int c = cbase + mm;
            if (r < M && c < N) {
                float v = acc[i];
                if (bias) v += bias[c];
                if (act)  v = silu_f(v);
                if (outF) outF[(size_t)r * N + c] = v;
                if (outB) outB[(size_t)r * N + c] = f2bf(v);
            }
        }
    };
    store_acc(c00, row0 + wr * 32,      col0 + wc * 32);
    store_acc(c01, row0 + wr * 32,      col0 + wc * 32 + 16);
    store_acc(c10, row0 + wr * 32 + 16, col0 + wc * 32);
    store_acc(c11, row0 + wr * 32 + 16, col0 + wc * 32 + 16);
}

// ---------------------------------------------------------------------------
// Fused edge-MLP stage 1: m1 = silu(e_in @ eW1 + eb1)
// e_in[e,:] = [h[row(e)], h[col(e)], radial(e), edge_attr(e)] (K=518, KP=544)
// row(e)=e/30, col(e)=(e/900)*30+e%30. h is bf16; phases 0/1 stage via async.
// ---------------------------------------------------------------------------
__global__ __launch_bounds__(256) void edge_gemm1_kernel(
    const unsigned short* __restrict__ hb,   // [N,256] bf16
    const float* __restrict__ radial,        // [E]
    const float* __restrict__ eattr,         // [E,5]
    const unsigned short* __restrict__ Wt,   // [256][544] bf16 transposed
    const float* __restrict__ bias,          // [256]
    unsigned short* __restrict__ outB,       // [E,256] bf16
    int E)
{
    const int Nn = 256, KP = 544;

    __shared__ __align__(16) unsigned short sA[BM][LDR];
    __shared__ __align__(16) unsigned short sBt[BN][LDR];

    const int tid  = threadIdx.x;
    const int lane = tid & 31;
    const int wv   = tid >> 5;
    const int wr   = wv & 3;
    const int wc   = wv >> 2;
    const int row0 = blockIdx.y * BM;
    const int col0 = blockIdx.x * BN;
    const int mm   = lane & 15;
    const int half = lane >> 4;

    const int ar   = tid >> 1;
    const int akb  = (tid & 1) << 4;
    const int e    = row0 + ar;                 // < E (E % 128 == 0)
    const int nrow = e / 30;
    const int ncol = (e / 900) * 30 + (e % 30);
    const int bn   = tid >> 2;
    const int bkc  = (tid & 3) << 3;

    v8f c00 = {}, c01 = {}, c10 = {}, c11 = {};

    auto stage_B = [&](int k0) {
        async_ld_b128(ldsoff(&sBt[bn][bkc]),
                      Wt + (size_t)(col0 + bn) * KP + k0 + bkc);
    };
    auto stage_A_async = [&](const unsigned short* src) {
        async_ld_b128(ldsoff(&sA[ar][akb]),     src + akb);
        async_ld_b128(ldsoff(&sA[ar][akb + 8]), src + akb + 8);
    };
    auto mma_step = [&]() {
        Frag a0, a1, b0, b1;
        a0.q[0] = *(const uint4*)&sA[wr * 32 + mm][8 * half];
        a0.q[1] = *(const uint4*)&sA[wr * 32 + mm][16 + 8 * half];
        a1.q[0] = *(const uint4*)&sA[wr * 32 + 16 + mm][8 * half];
        a1.q[1] = *(const uint4*)&sA[wr * 32 + 16 + mm][16 + 8 * half];
        b0.q[0] = *(const uint4*)&sBt[wc * 32 + mm][8 * half];
        b0.q[1] = *(const uint4*)&sBt[wc * 32 + mm][16 + 8 * half];
        b1.q[0] = *(const uint4*)&sBt[wc * 32 + 16 + mm][8 * half];
        b1.q[1] = *(const uint4*)&sBt[wc * 32 + 16 + mm][16 + 8 * half];
        c00 = wmma_bf16(a0, b0, c00);
        c01 = wmma_bf16(a0, b1, c01);
        c10 = wmma_bf16(a1, b0, c10);
        c11 = wmma_bf16(a1, b1, c11);
    };

    const unsigned short* hrow = hb + (size_t)nrow * 256;
    const unsigned short* hcol = hb + (size_t)ncol * 256;

    // phase 0: k in [0,256) -> h[row]
    for (int k0 = 0; k0 < 256; k0 += BK) {
        stage_A_async(hrow + k0);
        stage_B(k0);
        wait_async0();
        __syncthreads();
        mma_step();
        __syncthreads();
    }
    // phase 1: k in [256,512) -> h[col]
    for (int k0 = 256; k0 < 512; k0 += BK) {
        stage_A_async(hcol + (k0 - 256));
        stage_B(k0);
        wait_async0();
        __syncthreads();
        mma_step();
        __syncthreads();
    }
    // tail: k in [512,544) -> [radial, edge_attr, 0...]
    {
        const int k0 = 512;
        Frag p;
#pragma unroll
        for (int j = 0; j < 16; ++j) {
            const int gk = k0 + akb + j;
            float x = 0.0f;
            if (gk == 512)     x = radial[e];
            else if (gk < 518) x = eattr[(size_t)e * 5 + (gk - 513)];
            p.u[j] = f2bf(x);
        }
        *(uint4*)&sA[ar][akb]     = p.q[0];
        *(uint4*)&sA[ar][akb + 8] = p.q[1];
        stage_B(k0);
        wait_async0();
        __syncthreads();
        mma_step();
        __syncthreads();
    }

    auto store_acc = [&](const v8f& acc, int rbase, int cbase) {
#pragma unroll
        for (int i = 0; i < 8; ++i) {
            const int r = rbase + half * 8 + i;
            const int c = cbase + mm;
            float v = acc[i] + bias[c];
            outB[(size_t)r * Nn + c] = f2bf(silu_f(v));
        }
    };
    store_acc(c00, row0 + wr * 32,      col0 + wc * 32);
    store_acc(c01, row0 + wr * 32,      col0 + wc * 32 + 16);
    store_acc(c10, row0 + wr * 32 + 16, col0 + wc * 32);
    store_acc(c11, row0 + wr * 32 + 16, col0 + wc * 32 + 16);
}

// ---------------------------------------------------------------------------
// Elementwise / segment kernels (segments are contiguous runs of 30 edges)
// ---------------------------------------------------------------------------
__global__ void concat_z_label_kernel(const float* __restrict__ z,
                                      const float* __restrict__ label,
                                      float* __restrict__ zc, int Nn) {
    int i = blockIdx.x * blockDim.x + threadIdx.x;
    if (i >= Nn * 71) return;
    int n = i / 71, k = i % 71;
    zc[i] = (k < 64) ? z[n * 64 + k] : label[n * 7 + (k - 64)];
}

__global__ void radial_kernel(const float* __restrict__ coord,
                              float* __restrict__ cdiff,
                              float* __restrict__ rad, int E) {
    int e = blockIdx.x * blockDim.x + threadIdx.x;
    if (e >= E) return;
    int nr = e / 30;
    int nc = (e / 900) * 30 + (e % 30);
    float dx = coord[nr * 3 + 0] - coord[nc * 3 + 0];
    float dy = coord[nr * 3 + 1] - coord[nc * 3 + 1];
    float dz = coord[nr * 3 + 2] - coord[nc * 3 + 2];
    cdiff[e * 3 + 0] = dx;
    cdiff[e * 3 + 1] = dy;
    cdiff[e * 3 + 2] = dz;
    rad[e] = dx * dx + dy * dy + dz * dz;
}

__global__ void coord_update_kernel(const float* __restrict__ cdiff,
                                    const float* __restrict__ we,
                                    float* __restrict__ coord, int Nn) {
    int i = blockIdx.x * blockDim.x + threadIdx.x;  // over Nn*3
    if (i >= Nn * 3) return;
    int n = i / 3, d = i % 3;
    float s = 0.0f;
#pragma unroll 6
    for (int j = 0; j < 30; ++j) {
        int e = n * 30 + j;
        s += cdiff[e * 3 + d] * we[e];
    }
    coord[i] += s * (1.0f / 30.0f);
}

__global__ void agg_kernel(const unsigned short* __restrict__ m,
                           float* __restrict__ agg, int Nn) {
    int i = blockIdx.x * blockDim.x + threadIdx.x;  // over Nn*256
    if (i >= Nn * 256) return;
    int n = i >> 8, c = i & 255;
    float s = 0.0f;
#pragma unroll 6
    for (int j = 0; j < 30; ++j)
        s += bf2f(m[(size_t)(n * 30 + j) * 256 + c]);
    agg[i] = s;
}

// nin (bf16) = [h, agg, h]
__global__ void build_nin_kernel(const unsigned short* __restrict__ hb,
                                 const float* __restrict__ agg,
                                 unsigned short* __restrict__ nin, int Nn) {
    int i = blockIdx.x * blockDim.x + threadIdx.x;  // over Nn*768
    if (i >= Nn * 768) return;
    int n = i / 768, k = i % 768;
    unsigned short v;
    if (k < 256)       v = hb[n * 256 + k];
    else if (k < 512)  v = f2bf(agg[n * 256 + (k - 256)]);
    else               v = hb[n * 256 + (k - 512)];
    nin[i] = v;
}

__global__ void copy_f32_kernel(const float* __restrict__ src,
                                float* __restrict__ dst, int n) {
    int i = blockIdx.x * blockDim.x + threadIdx.x;
    if (i < n) dst[i] = src[i];
}

// ---------------------------------------------------------------------------
// Host orchestration
// ---------------------------------------------------------------------------
extern "C" void kernel_launch(void* const* d_in, const int* in_sizes, int n_in,
                              void* d_out, int out_size, void* d_ws, size_t ws_size,
                              hipStream_t stream) {
    const int B   = 128, PAD = 30;
    const int Nn  = B * PAD;          // 3840
    const int E   = B * PAD * PAD;    // 115200
    const int ZIN = 71;

    const float* z      = (const float*)d_in[0];
    const float* label  = (const float*)d_in[1];
    const float* emb_W  = (const float*)d_in[4];
    const float* emb_b  = (const float*)d_in[5];
    const float* rn_W1  = (const float*)d_in[6];
    const float* rn_b1  = (const float*)d_in[7];
    const float* rn_W2  = (const float*)d_in[8];
    const float* rn_b2  = (const float*)d_in[9];
    const float* rp_W1  = (const float*)d_in[10];
    const float* rp_b1  = (const float*)d_in[11];
    const float* rp_W2  = (const float*)d_in[12];
    const float* rp_b2  = (const float*)d_in[13];
    const float* re_W1  = (const float*)d_in[14];
    const float* re_b1  = (const float*)d_in[15];
    const float* re_W2  = (const float*)d_in[16];
    const float* re_b2  = (const float*)d_in[17];
    const float* an_W1  = (const float*)d_in[18];
    const float* an_b1  = (const float*)d_in[19];
    const float* an_W2  = (const float*)d_in[20];
    const float* an_b2  = (const float*)d_in[21];

    float* out       = (float*)d_out;
    float* out_h     = out;                       // [3840,5]
    float* out_coord = out_h + Nn * 5;            // [3840,3]
    float* out_eattr = out_coord + Nn * 3;        // [115200,5]
    float* out_atom  = out_eattr + (size_t)E * 5; // [128,30]

    char* wsp = (char*)d_ws;
    auto alloc = [&](size_t bytes) -> void* {
        void* p = (void*)wsp;
        wsp += (bytes + 255) & ~(size_t)255;
        return p;
    };
    float* zc    = (float*)alloc((size_t)Nn * ZIN * 4);
    float* coord = (float*)alloc((size_t)Nn * 3 * 4);
    float* t15   = (float*)alloc((size_t)Nn * 15 * 4);
    float* t60   = (float*)alloc((size_t)B * 60 * 4);
    float* cdiff = (float*)alloc((size_t)E * 3 * 4);
    float* rad   = (float*)alloc((size_t)E * 4);
    float* wE    = (float*)alloc((size_t)E * 4);
    float* agg   = (float*)alloc((size_t)Nn * 256 * 4);
    unsigned short* hb   = (unsigned short*)alloc((size_t)Nn * 256 * 2);
    unsigned short* ninb = (unsigned short*)alloc((size_t)Nn * 768 * 2);
    unsigned short* tnb  = (unsigned short*)alloc((size_t)Nn * 256 * 2);
    unsigned short* t128 = (unsigned short*)alloc((size_t)Nn * 128 * 2);
    unsigned short* m1   = (unsigned short*)alloc((size_t)E * 256 * 2);
    unsigned short* m2   = (unsigned short*)alloc((size_t)E * 256 * 2);

    // ---- weight prep: bf16 transpose with KP = ceil32(K) ----
    auto prep = [&](const float* W, int K, int N, int& KPo) -> unsigned short* {
        const int KP = (K + 31) & ~31;
        unsigned short* Wt = (unsigned short*)alloc((size_t)N * KP * 2);
        xpose_w_kernel<<<((N * KP) + 255) / 256, 256, 0, stream>>>(W, Wt, K, N, KP);
        KPo = KP;
        return Wt;
    };
    int kp_emb, kp_rn1, kp_rn2, kp_rp1, kp_rp2, kp_re1, kp_re2, kp_an1, kp_an2;
    unsigned short* emb_Wt = prep(emb_W, ZIN, 256, kp_emb);
    unsigned short* rn_W1t = prep(rn_W1, 256, 128, kp_rn1);
    unsigned short* rn_W2t = prep(rn_W2, 128, 5, kp_rn2);
    unsigned short* rp_W1t = prep(rp_W1, ZIN, 15, kp_rp1);
    unsigned short* rp_W2t = prep(rp_W2, 15, 3, kp_rp2);
    unsigned short* re_W1t = prep(re_W1, PAD * ZIN, 60, kp_re1);
    unsigned short* re_W2t = prep(re_W2, 60, PAD * PAD * 5, kp_re2);
    unsigned short* an_W1t = prep(an_W1, PAD * ZIN, 60, kp_an1);
    unsigned short* an_W2t = prep(an_W2, 60, PAD, kp_an2);
    unsigned short* eW1t[2], *eW2t[2], *nW1t[2], *nW2t[2], *cW1t[2], *cW2t[2];
    int kp_e1, kp_e2, kp_n1, kp_n2, kp_c1, kp_c2;
    for (int l = 0; l < 2; ++l) {
        const int base = 22 + 11 * l;
        eW1t[l] = prep((const float*)d_in[base + 0], 518, 256, kp_e1);
        eW2t[l] = prep((const float*)d_in[base + 2], 256, 256, kp_e2);
        nW1t[l] = prep((const float*)d_in[base + 4], 768, 256, kp_n1);
        nW2t[l] = prep((const float*)d_in[base + 6], 256, 256, kp_n2);
        cW1t[l] = prep((const float*)d_in[base + 8], 256, 256, kp_c1);
        cW2t[l] = prep((const float*)d_in[base + 10], 256, 1, kp_c2);
    }

    auto gemm_f32 = [&](const float* A, const unsigned short* Wt, int KP,
                        const float* bm, float* oF, unsigned short* oB,
                        int M, int N, int K, int act) {
        dim3 g((N + BN - 1) / BN, (M + BM - 1) / BM);
        gemm_kernel<0><<<g, 256, 0, stream>>>(A, Wt, KP, bm, oF, oB, M, N, K, act);
    };
    auto gemm_bf = [&](const unsigned short* A, const unsigned short* Wt, int KP,
                       const float* bm, float* oF, unsigned short* oB,
                       int M, int N, int K, int act) {
        dim3 g((N + BN - 1) / BN, (M + BM - 1) / BM);
        gemm_kernel<1><<<g, 256, 0, stream>>>(A, Wt, KP, bm, oF, oB, M, N, K, act);
    };

    // z_ = concat(z, label)
    concat_z_label_kernel<<<(Nn * ZIN + 255) / 256, 256, 0, stream>>>(z, label, zc, Nn);

    // atom_num = MLP(zg), zg = zc viewed [B, 2130]
    gemm_f32(zc, an_W1t, kp_an1, an_b1, t60, nullptr, B, 60, PAD * ZIN, 1);
    gemm_f32(t60, an_W2t, kp_an2, an_b2, out_atom, nullptr, B, PAD, 60, 0);

    // h = z_ @ emb_W + emb_b  (bf16 for all downstream consumers)
    gemm_f32(zc, emb_Wt, kp_emb, emb_b, nullptr, hb, Nn, 256, ZIN, 0);

    // coord = MLP(z_)
    gemm_f32(zc, rp_W1t, kp_rp1, rp_b1, t15, nullptr, Nn, 15, ZIN, 1);
    gemm_f32(t15, rp_W2t, kp_rp2, rp_b2, coord, nullptr, Nn, 3, 15, 0);

    // edge_attr = MLP(zg).reshape(E,5) straight into d_out slice
    gemm_f32(zc, re_W1t, kp_re1, re_b1, t60, nullptr, B, 60, PAD * ZIN, 1);
    gemm_f32(t60, re_W2t, kp_re2, re_b2, out_eattr, nullptr, B, PAD * PAD * 5, 60, 0);

    // EGCL layers
    for (int l = 0; l < 2; ++l) {
        const int base = 22 + 11 * l;
        const float* eb1 = (const float*)d_in[base + 1];
        const float* eb2 = (const float*)d_in[base + 3];
        const float* nb1 = (const float*)d_in[base + 5];
        const float* nb2 = (const float*)d_in[base + 7];
        const float* cb1 = (const float*)d_in[base + 9];

        radial_kernel<<<(E + 255) / 256, 256, 0, stream>>>(coord, cdiff, rad, E);

        {   // m = silu(e_in @ eW1 + eb1), fused gather
            dim3 g(256 / BN, E / BM);
            edge_gemm1_kernel<<<g, 256, 0, stream>>>(hb, rad, out_eattr,
                                                     eW1t[l], eb1, m1, E);
        }
        gemm_bf(m1, eW2t[l], kp_e2, eb2, nullptr, m2, E, 256, 256, 1);
        gemm_bf(m2, cW1t[l], kp_c1, cb1, nullptr, m1, E, 256, 256, 1);
        gemm_bf(m1, cW2t[l], kp_c2, nullptr, wE, nullptr, E, 1, 256, 0);

        coord_update_kernel<<<(Nn * 3 + 255) / 256, 256, 0, stream>>>(cdiff, wE, coord, Nn);

        agg_kernel<<<(Nn * 256 + 255) / 256, 256, 0, stream>>>(m2, agg, Nn);
        build_nin_kernel<<<(Nn * 768 + 255) / 256, 256, 0, stream>>>(hb, agg, ninb, Nn);
        gemm_bf(ninb, nW1t[l], kp_n1, nb1, nullptr, tnb, Nn, 256, 768, 1);
        gemm_bf(tnb, nW2t[l], kp_n2, nb2, nullptr, hb, Nn, 256, 256, 0);
    }

    // h_out = MLP(h)
    gemm_bf(hb, rn_W1t, kp_rn1, rn_b1, nullptr, t128, Nn, 128, 256, 1);
    gemm_bf(t128, rn_W2t, kp_rn2, rn_b2, out_h, nullptr, Nn, 5, 128, 0);

    copy_f32_kernel<<<(Nn * 3 + 255) / 256, 256, 0, stream>>>(coord, out_coord, Nn * 3);
}
